// Skip2Attention_37598143709787
// MI455X (gfx1250) — compile-verified, matrix operand order checked by reading, not run on hardware
//
#include <hip/hip_runtime.h>
#include <hip/hip_bf16.h>

typedef __bf16 bf16_t;
typedef __attribute__((ext_vector_type(8)))  bf16_t v8bf;
typedef __attribute__((ext_vector_type(16))) bf16_t v16bf;
typedef __attribute__((ext_vector_type(8)))  float  v8f;

#define DEV __device__ __forceinline__

// Problem constants (fixed by the reference setup)
constexpr int Bb = 4;            // batch
constexpr int T  = 16;           // ctx time steps
constexpr int T0 = 16;           // query time blocks
constexpr int Lq = 64;           // latent size
constexpr int C  = 512;          // channels
constexpr int H  = 8;            // heads
constexpr int HD = 64;           // head dim
constexpr int N  = T0 * Lq;      // 1024 query tokens
constexpr int TL = T  * Lq;      // 1024 ctx tokens
constexpr float SCALE = 0.125f;  // hd^-0.5
constexpr float NEGI  = -1e9f;

constexpr int BPAD = 72;         // padded LDS row stride (bf16 elems) for B tiles
constexpr int GTHREADS = 256;    // GEMM block: 8 waves, 128-row output tile

DEV v8f vzero8() {
  v8f z;
#pragma unroll
  for (int i = 0; i < 8; ++i) z[i] = 0.0f;
  return z;
}

DEV v8f wmma_bf16(v16bf a, v16bf b, v8f c) {
  // D = A(16x32 bf16) * B(32x16 bf16) + C(16x16 f32)
  return __builtin_amdgcn_wmma_f32_16x16x32_bf16(false, a, false, b, (short)0, c,
                                                 false, false);
}

// Async copy 16 bytes Global -> LDS (VGLOBAL GLOBAL_LOAD_ASYNC_TO_LDS_B128,
// tracked with ASYNCcnt).  LDS address = low 32 bits of the generic pointer.
DEV void async_copy_b128(void* lds_dst, const void* gsrc) {
  asm volatile("global_load_async_to_lds_b128 %0, %1, off"
               :: "v"((unsigned)(uintptr_t)lds_dst), "v"(gsrc)
               : "memory");
}
DEV void wait_async0() {
  asm volatile("s_wait_asynccnt 0x0" ::: "memory");
}

// Load a 16x32 bf16 A-fragment (also used for B-fragments given B^T row-major).
// base points at (row0, col0); ld = row stride in elements.
// Lane l: row = l%16; halves 0..7 = cols 8*(l/16)..+7, halves 8..15 = +16.
DEV v16bf frag_bf16(const bf16_t* base, int ld) {
  int lane = threadIdx.x & 31;
  const bf16_t* q = base + (lane & 15) * ld + ((lane >> 4) << 3);
  v8bf lo = *(const v8bf*)q;
  v8bf hi = *(const v8bf*)(q + 16);
  v16bf r;
#pragma unroll
  for (int i = 0; i < 8; ++i) { r[i] = lo[i]; r[i + 8] = hi[i]; }
  return r;
}

// Same fragment, but source is fp32 in global memory: load + convert to bf16.
DEV v16bf frag_f32(const float* base, int ld) {
  int lane = threadIdx.x & 31;
  const float* q = base + (lane & 15) * ld + ((lane >> 4) << 3);
  float4 a = *(const float4*)(q + 0);
  float4 b = *(const float4*)(q + 4);
  float4 c = *(const float4*)(q + 16);
  float4 d = *(const float4*)(q + 20);
  v16bf r;
  r[0]  = (bf16_t)a.x; r[1]  = (bf16_t)a.y; r[2]  = (bf16_t)a.z; r[3]  = (bf16_t)a.w;
  r[4]  = (bf16_t)b.x; r[5]  = (bf16_t)b.y; r[6]  = (bf16_t)b.z; r[7]  = (bf16_t)b.w;
  r[8]  = (bf16_t)c.x; r[9]  = (bf16_t)c.y; r[10] = (bf16_t)c.z; r[11] = (bf16_t)c.w;
  r[12] = (bf16_t)d.x; r[13] = (bf16_t)d.y; r[14] = (bf16_t)d.z; r[15] = (bf16_t)d.w;
  return r;
}

// Stage one 64(row) x 64(k) bf16 weight tile into LDS (padded rows) with
// async global->LDS copies: 256 threads x 2 x 16B = 8 KB.
DEV void stage_b(const bf16_t* wt, int n0, int kc, bf16_t* buf) {
  int tid = threadIdx.x;
#pragma unroll
  for (int j = 0; j < 2; ++j) {
    int e = tid + j * GTHREADS;     // 512 x 16B transfers
    int row = e >> 3;               // 8 x 16B per 128B (64-elem) row
    int colb = (e & 7) << 4;        // byte offset within row
    const char* src = (const char*)(wt + (size_t)(n0 + row) * C + kc * 64) + colb;
    char* dst = (char*)buf + row * (BPAD * 2) + colb;
    async_copy_b128(dst, src);
  }
}

// Core GEMM tile: A rows from global (fp32 or bf16), B (= W^T 64-col slice)
// double-buffered through LDS via async copies.  acc[c] accumulates the
// wave's 16x64 strip.  Abase points at (row m0_of_wave, col 0), lda = C.
template<bool AF32>
DEV void gemm_core(const char* Abase, const bf16_t* wt, int n0,
                   bf16_t* bs0, bf16_t* bs1, v8f acc[4]) {
  stage_b(wt, n0, 0, bs0);
  for (int kc = 0; kc < C / 64; ++kc) {
    bf16_t* cur = (kc & 1) ? bs1 : bs0;
    bf16_t* nxt = (kc & 1) ? bs0 : bs1;
    wait_async0();        // our async copies for `cur` have landed
    __syncthreads();      // everyone's copies landed; other buffer free
    if (kc + 1 < C / 64) {
      stage_b(wt, n0, kc + 1, nxt);
      // speculative prefetch of next A chunk (global_prefetch_b8)
      __builtin_prefetch(Abase + (size_t)(kc + 1) * 64 * (AF32 ? 4 : 2), 0, 0);
    }
#pragma unroll
    for (int s = 0; s < 2; ++s) {
      int kk = kc * 64 + s * 32;
      v16bf a;
      if constexpr (AF32) a = frag_f32((const float*)Abase + kk, C);
      else                a = frag_bf16((const bf16_t*)Abase + kk, C);
#pragma unroll
      for (int c = 0; c < 4; ++c) {
        v16bf b = frag_bf16(cur + (16 * c) * BPAD + s * 32, BPAD);
        acc[c] = wmma_bf16(a, b, acc[c]);
      }
    }
  }
}

// ---------------------------------------------------------------------------
// Weight transpose + fp32 -> bf16 convert:  dst[cols][rows] = src[rows][cols]
// ---------------------------------------------------------------------------
__global__ void k_transpose_cvt(const float* __restrict__ src,
                                bf16_t* __restrict__ dst, int rows, int cols) {
  int idx = blockIdx.x * blockDim.x + threadIdx.x;
  if (idx >= rows * cols) return;
  int i = idx / cols, j = idx - i * cols;
  dst[j * rows + i] = (bf16_t)src[idx];
}

// ---------------------------------------------------------------------------
// qkv = x @ W_qkv   (M=4096, K=512, Nout=1536), scattered into
//   q[b][h][n][d], k[b][h][n][d], vT[b][h][d][n]   (all bf16)
// ---------------------------------------------------------------------------
__global__ void __launch_bounds__(GTHREADS)
k_qkv(const float* __restrict__ x, const bf16_t* __restrict__ wqt,
      bf16_t* __restrict__ qws, bf16_t* __restrict__ kws,
      bf16_t* __restrict__ vtws) {
  __shared__ __align__(16) bf16_t bsm[2][64 * BPAD];
  int lane = threadIdx.x & 31, wave = threadIdx.x >> 5;
  int m0 = blockIdx.x * 128 + wave * 16;
  int n0 = blockIdx.y * 64;
  v8f acc[4];
#pragma unroll
  for (int c = 0; c < 4; ++c) acc[c] = vzero8();

  gemm_core<true>((const char*)(x + (size_t)m0 * C), wqt, n0,
                  &bsm[0][0], &bsm[1][0], acc);

  int which = n0 >> 9;          // 0=q, 1=k, 2=v
  int h = (n0 & 511) >> 6;      // head (n0 is 64-aligned -> single head)
#pragma unroll
  for (int c = 0; c < 4; ++c) {
    int d = 16 * c + (lane & 15);
#pragma unroll
    for (int i = 0; i < 8; ++i) {
      int m = m0 + i + ((lane >> 4) << 3);
      int bb = m >> 10, nn = m & 1023;
      bf16_t bv = (bf16_t)acc[c][i];
      if (which == 0)      qws[((bb * H + h) * N + nn) * HD + d] = bv;
      else if (which == 1) kws[((bb * H + h) * N + nn) * HD + d] = bv;
      else                 vtws[((bb * H + h) * HD + d) * N + nn] = bv;
    }
  }
}

// ---------------------------------------------------------------------------
// k_ctx = dx_ctx @ W_k  (M=65536, K=512, Nout=512) -> kctx[b][h][t0][t*L+l][d]
// ---------------------------------------------------------------------------
__global__ void __launch_bounds__(GTHREADS)
k_kctx(const float* __restrict__ dx, const bf16_t* __restrict__ wkt,
       bf16_t* __restrict__ kctx) {
  __shared__ __align__(16) bf16_t bsm[2][64 * BPAD];
  int lane = threadIdx.x & 31, wave = threadIdx.x >> 5;
  int m0 = blockIdx.x * 128 + wave * 16;
  int n0 = blockIdx.y * 64;
  v8f acc[4];
#pragma unroll
  for (int c = 0; c < 4; ++c) acc[c] = vzero8();

  gemm_core<true>((const char*)(dx + (size_t)m0 * C), wkt, n0,
                  &bsm[0][0], &bsm[1][0], acc);

  int h = n0 >> 6;
#pragma unroll
  for (int c = 0; c < 4; ++c) {
    int d = 16 * c + (lane & 15);
#pragma unroll
    for (int i = 0; i < 8; ++i) {
      int m = m0 + i + ((lane >> 4) << 3);
      int bb = m >> 14;             // / (T*N)
      int tt = (m >> 10) & 15;      // t
      int nn = m & 1023;            // within N
      int t0i = nn >> 6, li = nn & 63;
      size_t idx = ((size_t)((bb * H + h) * T0 + t0i) * TL + (tt * Lq + li)) * HD + d;
      kctx[idx] = (bf16_t)acc[c][i];
    }
  }
}

// ---------------------------------------------------------------------------
// v_ctx = x_ctx @ W_v  (M=4096, K=512, Nout=512) -> vTctx[b][h][d][kk]
// ---------------------------------------------------------------------------
__global__ void __launch_bounds__(GTHREADS)
k_vctx(const float* __restrict__ xc, const bf16_t* __restrict__ wvt,
       bf16_t* __restrict__ vtctx) {
  __shared__ __align__(16) bf16_t bsm[2][64 * BPAD];
  int lane = threadIdx.x & 31, wave = threadIdx.x >> 5;
  int m0 = blockIdx.x * 128 + wave * 16;
  int n0 = blockIdx.y * 64;
  v8f acc[4];
#pragma unroll
  for (int c = 0; c < 4; ++c) acc[c] = vzero8();

  gemm_core<true>((const char*)(xc + (size_t)m0 * C), wvt, n0,
                  &bsm[0][0], &bsm[1][0], acc);

  int h = n0 >> 6;
#pragma unroll
  for (int c = 0; c < 4; ++c) {
    int d = 16 * c + (lane & 15);
#pragma unroll
    for (int i = 0; i < 8; ++i) {
      int m = m0 + i + ((lane >> 4) << 3);
      int bb = m >> 10, kkix = m & 1023;
      vtctx[((bb * H + h) * HD + d) * TL + kkix] = (bf16_t)acc[c][i];
    }
  }
}

// ---------------------------------------------------------------------------
// Flash attention per (b, h, t0): 4 waves, each owns a 16-row query tile.
// 34 chunks x 32 keys: 32 ctx chunks + 2 self chunks, online softmax,
// P tile bounced through per-wave LDS to convert D-layout -> A-layout.
// ---------------------------------------------------------------------------
__global__ void __launch_bounds__(128)
k_attn(const bf16_t* __restrict__ qws, const bf16_t* __restrict__ kws,
       const bf16_t* __restrict__ vtws, const bf16_t* __restrict__ kctx,
       const bf16_t* __restrict__ vtctx, const unsigned char* __restrict__ mask,
       bf16_t* __restrict__ aout) {
  __shared__ __align__(16) bf16_t pbuf[4][16 * 32];
  int lane = threadIdx.x & 31, wave = threadIdx.x >> 5;
  int t0 = blockIdx.x, h = blockIdx.y, b = blockIdx.z;
  int m0 = wave * 16;

  const bf16_t* qblk    = qws  + (((b * H + h) * N + t0 * Lq + m0)) * HD;
  const bf16_t* kself   = kws  + (((b * H + h) * N + t0 * Lq)) * HD;
  const bf16_t* vtself  = vtws + ((size_t)(b * H + h) * HD) * N + t0 * Lq;
  const bf16_t* kctxblk = kctx + ((size_t)((b * H + h) * T0 + t0) * TL) * HD;
  const bf16_t* vtcblk  = vtctx + ((size_t)(b * H + h) * HD) * TL;
  bf16_t* pb = &pbuf[wave][0];

  v16bf qa0 = frag_bf16(qblk + 0, HD);
  v16bf qa1 = frag_bf16(qblk + 32, HD);

  float mrow[8], lrow[8], sf[8];
#pragma unroll
  for (int i = 0; i < 8; ++i) { mrow[i] = -1e30f; lrow[i] = 0.0f; }
  v8f o[4];
#pragma unroll
  for (int dd = 0; dd < 4; ++dd) o[dd] = vzero8();

  for (int ci = 0; ci < 34; ++ci) {
    const bf16_t *kp, *vp;
    bool ok;
    if (ci < 32) {                       // cross-attention keys
      int kk0 = ci * 32;
      kp = kctxblk + kk0 * HD;
      vp = vtcblk + kk0;
      ok = mask[b * T + (kk0 >> 6)] != 0;   // 32-chunk lies in one t
    } else {                             // block-diagonal self keys
      int s0i = (ci - 32) * 32;
      kp = kself + s0i * HD;
      vp = vtself + s0i;
      ok = true;
    }

    // S = Q @ K^T over this 32-key chunk (two 16-col D fragments).
    v16bf kb0 = frag_bf16(kp, HD);
    v16bf kb1 = frag_bf16(kp + 32, HD);
    v16bf kb2 = frag_bf16(kp + 16 * HD, HD);
    v16bf kb3 = frag_bf16(kp + 16 * HD + 32, HD);
    v8f s0 = vzero8(), s1 = vzero8();
    s0 = wmma_bf16(qa0, kb0, s0);
    s0 = wmma_bf16(qa1, kb1, s0);
    s1 = wmma_bf16(qa0, kb2, s1);
    s1 = wmma_bf16(qa1, kb3, s1);

    // Preload V fragments early (latency hidden behind softmax VALU work).
    v16bf vb[4];
#pragma unroll
    for (int dd = 0; dd < 4; ++dd)
      vb[dd] = frag_bf16(vp + dd * 16 * 1024, 1024);

    // Online softmax update (row reductions across 16-lane halves)
#pragma unroll
    for (int i = 0; i < 8; ++i) {
      float a0 = ok ? s0[i] * SCALE : NEGI;
      float a1 = ok ? s1[i] * SCALE : NEGI;
      float tm = fmaxf(a0, a1);
      tm = fmaxf(tm, __shfl_xor(tm, 1));
      tm = fmaxf(tm, __shfl_xor(tm, 2));
      tm = fmaxf(tm, __shfl_xor(tm, 4));
      tm = fmaxf(tm, __shfl_xor(tm, 8));
      float mn = fmaxf(mrow[i], tm);
      float f  = __expf(mrow[i] - mn);
      float p0 = __expf(a0 - mn);
      float p1 = __expf(a1 - mn);
      float rs = p0 + p1;
      rs += __shfl_xor(rs, 1);
      rs += __shfl_xor(rs, 2);
      rs += __shfl_xor(rs, 4);
      rs += __shfl_xor(rs, 8);
      lrow[i] = lrow[i] * f + rs;
      mrow[i] = mn;
      sf[i]   = f;
      int rt = i + ((lane >> 4) << 3);
      pb[rt * 32 + (lane & 15)]      = (bf16_t)p0;
      pb[rt * 32 + 16 + (lane & 15)] = (bf16_t)p1;
    }
#pragma unroll
    for (int dd = 0; dd < 4; ++dd)
#pragma unroll
      for (int i = 0; i < 8; ++i) o[dd][i] *= sf[i];

    // O += P(16x32) @ V(32x64)
    v16bf pf = frag_bf16(pb, 32);
#pragma unroll
    for (int dd = 0; dd < 4; ++dd)
      o[dd] = wmma_bf16(pf, vb[dd], o[dd]);
  }

  float inv[8];
#pragma unroll
  for (int i = 0; i < 8; ++i) inv[i] = 1.0f / lrow[i];
#pragma unroll
  for (int dd = 0; dd < 4; ++dd)
#pragma unroll
    for (int i = 0; i < 8; ++i) {
      int row = t0 * Lq + m0 + i + ((lane >> 4) << 3);
      int cc  = h * HD + dd * 16 + (lane & 15);
      aout[((size_t)b * N + row) * C + cc] = (bf16_t)(o[dd][i] * inv[i]);
    }
}

// ---------------------------------------------------------------------------
// out = aout @ W_proj + b_proj   (M=4096, K=512, Nout=512), fp32 output
// ---------------------------------------------------------------------------
__global__ void __launch_bounds__(GTHREADS)
k_out(const bf16_t* __restrict__ aout, const bf16_t* __restrict__ wpt,
      const float* __restrict__ bias, float* __restrict__ out) {
  __shared__ __align__(16) bf16_t bsm[2][64 * BPAD];
  int lane = threadIdx.x & 31, wave = threadIdx.x >> 5;
  int m0 = blockIdx.x * 128 + wave * 16;
  int n0 = blockIdx.y * 64;
  v8f acc[4];
#pragma unroll
  for (int c = 0; c < 4; ++c) acc[c] = vzero8();

  gemm_core<false>((const char*)(aout + (size_t)m0 * C), wpt, n0,
                   &bsm[0][0], &bsm[1][0], acc);

#pragma unroll
  for (int c = 0; c < 4; ++c) {
    int j = n0 + 16 * c + (lane & 15);
    float bj = bias[j];
#pragma unroll
    for (int i = 0; i < 8; ++i) {
      int m = m0 + i + ((lane >> 4) << 3);
      out[(size_t)m * C + j] = acc[c][i] + bj;
    }
  }
}

// ---------------------------------------------------------------------------
extern "C" void kernel_launch(void* const* d_in, const int* in_sizes, int n_in,
                              void* d_out, int out_size, void* d_ws, size_t ws_size,
                              hipStream_t stream) {
  (void)in_sizes; (void)n_in; (void)out_size; (void)ws_size;
  const float* x    = (const float*)d_in[0];
  const float* xctx = (const float*)d_in[1];
  const float* dx   = (const float*)d_in[2];
  const unsigned char* mask = (const unsigned char*)d_in[3];
  const float* Wqkv = (const float*)d_in[4];
  const float* Wk   = (const float*)d_in[5];
  const float* Wv   = (const float*)d_in[6];
  const float* Wp   = (const float*)d_in[7];
  const float* bp   = (const float*)d_in[8];
  float* out = (float*)d_out;

  char* ws = (char*)d_ws;
  size_t off = 0;
  auto alloc = [&](size_t elems) {
    bf16_t* p = (bf16_t*)(ws + off);
    off += elems * sizeof(bf16_t);
    return p;
  };
  bf16_t* wqt    = alloc((size_t)C * 3 * C);        // W_qkv^T  [1536][512]
  bf16_t* wkt    = alloc((size_t)C * C);            // W_k^T
  bf16_t* wvt    = alloc((size_t)C * C);            // W_v^T
  bf16_t* wpt    = alloc((size_t)C * C);            // W_proj^T
  bf16_t* qws    = alloc((size_t)Bb * H * N * HD);  // q [b][h][n][d]
  bf16_t* kws    = alloc((size_t)Bb * H * N * HD);  // k [b][h][n][d]
  bf16_t* vtws   = alloc((size_t)Bb * H * HD * N);  // v^T [b][h][d][n]
  bf16_t* kctxws = alloc((size_t)Bb * H * T0 * TL * HD); // [b][h][t0][kk][d]
  bf16_t* vtcws  = alloc((size_t)Bb * H * HD * TL); // vctx^T [b][h][d][kk]
  bf16_t* aoutws = alloc((size_t)Bb * N * C);       // pre-projection output

  // Weights: transpose + convert to bf16
  k_transpose_cvt<<<(C * 3 * C + 255) / 256, 256, 0, stream>>>(Wqkv, wqt, C, 3 * C);
  k_transpose_cvt<<<(C * C + 255) / 256, 256, 0, stream>>>(Wk, wkt, C, C);
  k_transpose_cvt<<<(C * C + 255) / 256, 256, 0, stream>>>(Wv, wvt, C, C);
  k_transpose_cvt<<<(C * C + 255) / 256, 256, 0, stream>>>(Wp, wpt, C, C);

  // Projections (WMMA + async LDS-staged weights)
  k_qkv <<<dim3((Bb * N) / 128, (3 * C) / 64), GTHREADS, 0, stream>>>(x, wqt, qws, kws, vtws);
  k_kctx<<<dim3((Bb * T * N) / 128, C / 64),   GTHREADS, 0, stream>>>(dx, wkt, kctxws);
  k_vctx<<<dim3((Bb * TL) / 128, C / 64),      GTHREADS, 0, stream>>>(xctx, wvt, vtcws);

  // Flash attention over merged (ctx + self) keys
  k_attn<<<dim3(T0, H, Bb), 128, 0, stream>>>(qws, kws, vtws, kctxws, vtcws, mask, aoutws);

  // Output projection + bias
  k_out<<<dim3((Bb * N) / 128, C / 64), GTHREADS, 0, stream>>>(aoutws, wpt, bp, out);
}
